// GraphAttentionEmbedding_43138651521154
// MI455X (gfx1250) — matrix operand order repaired and checked by reading, not compiled, not run on hardware
//
#include <hip/hip_runtime.h>

// ---------------------------------------------------------------------------
// GraphAttentionEmbedding (TGN TransformerConv) for gfx1250 / MI455X.
// bf16 WMMA (v_wmma_f32_16x16x32_bf16) for node QKV/skip GEMMs and the
// edge_attr @ We GEMM; streaming fused passes for time-encoding, segment
// softmax (encoded atomicMax + atomicAdd), and the weighted scatter.
// ---------------------------------------------------------------------------

typedef __attribute__((ext_vector_type(16))) __bf16 v16bf;
typedef __attribute__((ext_vector_type(8)))  float  v8f;

#define HC   128          // heads * per-head channels
#define EDIM 256          // msg + time encoding
#define SCALE 0.125f      // 1/sqrt(64)

static __device__ __forceinline__ long long llmin(long long a, long long b) {
    return a < b ? a : b;
}

// order-preserving encoding of float for unsigned atomicMax
static __device__ __forceinline__ unsigned int enc_f32(float f) {
    unsigned int u = __float_as_uint(f);
    return (u & 0x80000000u) ? ~u : (u | 0x80000000u);
}
static __device__ __forceinline__ float dec_f32(unsigned int u) {
    return __uint_as_float((u & 0x80000000u) ? (u & 0x7fffffffu) : ~u);
}
#define ENC_NEG_INF 0x007FFFFFu   // enc(-inf)

// ---------------------------------------------------------------------------
// Pack weights f32 -> bf16 in WMMA B-fragment layout.
// Block (kb,nb): lane<16: N=n0+lane, K=k0..k0+7,k0+16..k0+23 ; lane>=16: N=n0+lane-16,
// K=k0+8..k0+15,k0+24..k0+31 (ISA table order). 512 bf16 per block.
// packed block index: wBase[w] + kb*8 + nb ; wBase = {0,32,64,96,128}.
// ---------------------------------------------------------------------------
__global__ void pack_weights(const float* __restrict__ Wq,
                             const float* __restrict__ Wk,
                             const float* __restrict__ Wv,
                             const float* __restrict__ Wskip,
                             const float* __restrict__ We,
                             __bf16* __restrict__ packed) {
    int w  = blockIdx.y;              // 0..4
    int kb = blockIdx.x;              // 0..7 (only 0..3 valid for w<4)
    int nKB = (w == 4) ? 8 : 4;
    if (kb >= nKB) return;
    const float* W = (w == 0) ? Wq : (w == 1) ? Wk : (w == 2) ? Wv
                   : (w == 3) ? Wskip : We;
    int wBase = (w < 4) ? (w * 32) : 128;

    int tid  = threadIdx.x;           // 256 threads: nb(8) x lane(32)
    int nb   = tid >> 5;
    int lane = tid & 31;
    int k0   = kb * 32 + ((lane >= 16) ? 8 : 0);
    int n    = nb * 16 + (lane & 15);
    __bf16* dst = packed + (size_t)(wBase + kb * 8 + nb) * 512 + lane * 16;
#pragma unroll
    for (int i = 0; i < 16; ++i)
        dst[i] = (__bf16)W[(size_t)(k0 + i + ((i >= 8) ? 8 : 0)) * HC + n];
}

__global__ void init_nodes(unsigned int* __restrict__ m_enc,
                           float* __restrict__ denom, int n2) {
    int i = blockIdx.x * blockDim.x + threadIdx.x;
    if (i < n2) { m_enc[i] = ENC_NEG_INF; denom[i] = 0.0f; }
}

// ---------------------------------------------------------------------------
// Node GEMMs: q/k/v = x@W + b  (to ws), out = x@Wskip + bskip (to d_out).
// One wave per 16-row tile; A per ISA 16-bit 16x32 layout:
//   lane<16:  elems 0..7 -> K=k0..k0+7,   elems 8..15 -> K=k0+16..k0+23
//   lane>=16: elems 0..7 -> K=k0+8..k0+15,elems 8..15 -> K=k0+24..k0+31
// ---------------------------------------------------------------------------
#define NODE_WAVES 8
__global__ void node_gemm(const float* __restrict__ x,
                          const __bf16* __restrict__ packed,
                          const float* __restrict__ bq,
                          const float* __restrict__ bk,
                          const float* __restrict__ bv,
                          const float* __restrict__ bskip,
                          float* __restrict__ q, float* __restrict__ k,
                          float* __restrict__ v, float* __restrict__ out,
                          int Nn) {
    int lane = threadIdx.x & 31;
    int wave = threadIdx.x >> 5;
    int m0 = (blockIdx.x * NODE_WAVES + wave) * 16;
    if (m0 >= Nn) return;                       // no barriers in this kernel
    const bool fullTile = (m0 + 16 <= Nn);      // wave-uniform fast path
    int mrow = m0 + (lane & 15);
    int mclamp = mrow < Nn ? mrow : (Nn - 1);
    const float* xrow = x + (size_t)mclamp * HC;

    int koff = (lane >= 16) ? 8 : 0;
    v16bf a[4];
#pragma unroll
    for (int ks = 0; ks < 4; ++ks) {
        int kbase = ks * 32 + koff;
#pragma unroll
        for (int i = 0; i < 8; ++i) a[ks][i]     = (__bf16)xrow[kbase + i];
#pragma unroll
        for (int i = 0; i < 8; ++i) a[ks][8 + i] = (__bf16)xrow[kbase + 16 + i];
    }

    const float* biases[4] = {bq, bk, bv, bskip};
    float*       dsts[4]   = {q, k, v, out};
    int Mbase = (lane >= 16) ? 8 : 0;

    for (int w = 0; w < 4; ++w) {
        float* dptr = dsts[w];
#pragma unroll
        for (int nb = 0; nb < 8; ++nb) {
            float bb = biases[w][nb * 16 + (lane & 15)];
            v8f acc = {};
#pragma unroll
            for (int ks = 0; ks < 4; ++ks) {
                v16bf bfrag = *(const v16bf*)(packed +
                    (size_t)(w * 32 + ks * 8 + nb) * 512 + lane * 16);
                acc = __builtin_amdgcn_wmma_f32_16x16x32_bf16(
                    false, a[ks], false, bfrag, (short)0, acc, false, false);
            }
            int Ncol = nb * 16 + (lane & 15);
            float* dbase = dptr + (size_t)(m0 + Mbase) * HC + Ncol;
            if (fullTile) {
#pragma unroll
                for (int i = 0; i < 8; ++i)
                    dbase[(size_t)i * HC] = acc[i] + bb;
            } else {
#pragma unroll
                for (int i = 0; i < 8; ++i) {
                    int row = m0 + Mbase + i;
                    if (row < Nn) dptr[(size_t)row * HC + Ncol] = acc[i] + bb;
                }
            }
        }
    }
}

// ---------------------------------------------------------------------------
// Edge pass 1: e = [cos(rel_t*w+b) | msg] @ We  (A built in registers, never
// materialized in HBM). Stores e; stages tile in LDS; per-lane (edge,head)
// logit = SCALE * <q[dst], k[src]+e>; encoded atomicMax segment-max.
// ---------------------------------------------------------------------------
#define E1_WAVES 4
__global__ void edge_pass1(const float* __restrict__ last_update,
                           const long long* __restrict__ ei,   // [2,E]
                           const float* __restrict__ t,
                           const float* __restrict__ msg,
                           const float* __restrict__ time_w,
                           const float* __restrict__ time_b,
                           const __bf16* __restrict__ packed,
                           const float* __restrict__ q,
                           const float* __restrict__ k,
                           float* __restrict__ e_out,
                           float* __restrict__ logits,
                           unsigned int* __restrict__ m_enc,
                           long long E_) {
    __shared__ float lds_e[E1_WAVES][16][HC];   // 32 KB
    int lane = threadIdx.x & 31;
    int wave = threadIdx.x >> 5;
    long long e0 = ((long long)blockIdx.x * E1_WAVES + wave) * 16;
    const bool fullTile = (e0 + 16 <= E_);      // wave-uniform fast path

    long long myEdge = llmin(e0 + (lane & 15), E_ - 1);
    long long src0 = ei[myEdge];
    float relt = t[myEdge] - last_update[src0];

    int koff = (lane >= 16) ? 8 : 0;
    v16bf a[8];                                 // 16 edges x 256 K, bf16
#pragma unroll
    for (int ks = 0; ks < 8; ++ks) {
        int kbase = ks * 32 + koff;
#pragma unroll
        for (int half = 0; half < 2; ++half) {
            int kk0 = kbase + half * 16;
#pragma unroll
            for (int i = 0; i < 8; ++i) {
                int K = kk0 + i;
                float val = (K < 128)
                    ? __cosf(relt * time_w[K] + time_b[K])
                    : msg[myEdge * 128 + (K - 128)];
                a[ks][half * 8 + i] = (__bf16)val;
            }
        }
    }

    int Mbase = (lane >= 16) ? 8 : 0;
#pragma unroll
    for (int nb = 0; nb < 8; ++nb) {
        v8f acc = {};
#pragma unroll
        for (int ks = 0; ks < 8; ++ks) {
            v16bf bfrag = *(const v16bf*)(packed +
                (size_t)(128 + ks * 8 + nb) * 512 + lane * 16);
            acc = __builtin_amdgcn_wmma_f32_16x16x32_bf16(
                false, a[ks], false, bfrag, (short)0, acc, false, false);
        }
        int Ncol = nb * 16 + (lane & 15);
        float* ebase = e_out + (e0 + Mbase) * HC + Ncol;
        if (fullTile) {
#pragma unroll
            for (int i = 0; i < 8; ++i) {
                lds_e[wave][Mbase + i][Ncol] = acc[i];
                ebase[(size_t)i * HC] = acc[i];
            }
        } else {
#pragma unroll
            for (int i = 0; i < 8; ++i) {
                int M = Mbase + i;
                lds_e[wave][M][Ncol] = acc[i];
                long long erow = e0 + M;
                if (erow < E_) e_out[erow * HC + Ncol] = acc[i];
            }
        }
    }
    __syncthreads();

    // lane -> (edge_local, head); 16 edges x 2 heads == 32 lanes
    int el = lane >> 1, h = lane & 1;
    long long edge = llmin(e0 + el, E_ - 1);
    long long s2 = ei[edge];
    long long d2 = ei[E_ + edge];
    const float* qrow = q + (size_t)d2 * HC + h * 64;
    const float* krow = k + (size_t)s2 * HC + h * 64;
    float acc = 0.0f;
#pragma unroll 4
    for (int c = 0; c < 64; c += 4) {
        float4 qv = *(const float4*)(qrow + c);
        float4 kv = *(const float4*)(krow + c);
        acc += qv.x * (kv.x + lds_e[wave][el][h * 64 + c + 0]);
        acc += qv.y * (kv.y + lds_e[wave][el][h * 64 + c + 1]);
        acc += qv.z * (kv.z + lds_e[wave][el][h * 64 + c + 2]);
        acc += qv.w * (kv.w + lds_e[wave][el][h * 64 + c + 3]);
    }
    float logit = acc * SCALE;
    if (e0 + el < E_) {
        logits[(e0 + el) * 2 + h] = logit;
        atomicMax(&m_enc[d2 * 2 + h], enc_f32(logit));
    }
}

// ---------------------------------------------------------------------------
// Edge pass 2: p = exp(logit - m[dst]); denom[dst] += p.
// ---------------------------------------------------------------------------
__global__ void edge_pass2(const long long* __restrict__ ei,
                           const float* __restrict__ logits,
                           const unsigned int* __restrict__ m_enc,
                           float* __restrict__ p, float* __restrict__ denom,
                           long long E_) {
    long long i = (long long)blockIdx.x * blockDim.x + threadIdx.x;
    if (i >= E_ * 2) return;
    long long edge = i >> 1; int h = (int)(i & 1);
    long long d = ei[E_ + edge];
    float m = dec_f32(m_enc[d * 2 + h]);
    float pv = expf(logits[i] - m);
    p[i] = pv;
    atomicAdd(&denom[d * 2 + h], pv);
}

// ---------------------------------------------------------------------------
// Edge pass 3: out[dst] += (v[src] + e) * (p / (denom[dst] + eps)).
// One thread per (edge, 4-channel group); f32 atomic adds hit L2-resident out.
// ---------------------------------------------------------------------------
__global__ void edge_pass3(const long long* __restrict__ ei,
                           const float* __restrict__ e_ws,
                           const float* __restrict__ v_ws,
                           const float* __restrict__ p,
                           const float* __restrict__ denom,
                           float* __restrict__ out, long long E_) {
    long long i = (long long)blockIdx.x * blockDim.x + threadIdx.x;
    if (i >= E_ * 32) return;
    long long edge = i >> 5;
    int c4 = (int)(i & 31) * 4;
    int h = c4 >> 6;
    long long s = ei[edge];
    long long d = ei[E_ + edge];
    float alpha = p[edge * 2 + h] / (denom[d * 2 + h] + 1e-16f);
    float4 ev = *(const float4*)(e_ws + edge * HC + c4);
    float4 vv = *(const float4*)(v_ws + (size_t)s * HC + c4);
    float* ob = out + (size_t)d * HC + c4;
    atomicAdd(ob + 0, (vv.x + ev.x) * alpha);
    atomicAdd(ob + 1, (vv.y + ev.y) * alpha);
    atomicAdd(ob + 2, (vv.z + ev.z) * alpha);
    atomicAdd(ob + 3, (vv.w + ev.w) * alpha);
}

// ---------------------------------------------------------------------------
extern "C" void kernel_launch(void* const* d_in, const int* in_sizes, int n_in,
                              void* d_out, int out_size, void* d_ws, size_t ws_size,
                              hipStream_t stream) {
    const float*     x     = (const float*)d_in[0];
    const float*     lastu = (const float*)d_in[1];
    const long long* ei    = (const long long*)d_in[2];   // int64 [2,E]
    const float*     t     = (const float*)d_in[3];
    const float*     msg   = (const float*)d_in[4];
    const float*     tw    = (const float*)d_in[5];
    const float*     tb    = (const float*)d_in[6];
    const float*     Wq    = (const float*)d_in[7];
    const float*     bq    = (const float*)d_in[8];
    const float*     Wk    = (const float*)d_in[9];
    const float*     bk    = (const float*)d_in[10];
    const float*     Wv    = (const float*)d_in[11];
    const float*     bv    = (const float*)d_in[12];
    const float*     We    = (const float*)d_in[13];
    const float*     Wsk   = (const float*)d_in[14];
    const float*     bsk   = (const float*)d_in[15];

    const int       N = in_sizes[1];
    const long long E = in_sizes[3];

    // workspace layout (256B aligned chunks)
    char* w = (char*)d_ws;
    size_t off = 0;
    auto take = [&](size_t bytes) { char* p = w + off;
                                    off += (bytes + 255) & ~(size_t)255;
                                    return p; };
    __bf16*       packed = (__bf16*)      take((size_t)192 * 512 * 2);
    float*        qws    = (float*)       take((size_t)N * HC * 4);
    float*        kws    = (float*)       take((size_t)N * HC * 4);
    float*        vws    = (float*)       take((size_t)N * HC * 4);
    float*        ews    = (float*)       take((size_t)E * HC * 4);
    float*        logits = (float*)       take((size_t)E * 2 * 4);
    float*        pbuf   = (float*)       take((size_t)E * 2 * 4);
    unsigned int* m_enc  = (unsigned int*)take((size_t)N * 2 * 4);
    float*        denom  = (float*)       take((size_t)N * 2 * 4);
    float*        out    = (float*)d_out;

    pack_weights<<<dim3(8, 5), 256, 0, stream>>>(Wq, Wk, Wv, Wsk, We, packed);

    {
        int n2 = N * 2;
        init_nodes<<<(n2 + 255) / 256, 256, 0, stream>>>(m_enc, denom, n2);
    }
    {
        int rowsPerBlock = NODE_WAVES * 16;
        int blocks = (N + rowsPerBlock - 1) / rowsPerBlock;
        node_gemm<<<blocks, NODE_WAVES * 32, 0, stream>>>(
            x, packed, bq, bk, bv, bsk, qws, kws, vws, out, N);
    }
    {
        long long edgesPerBlock = E1_WAVES * 16;
        long long blocks = (E + edgesPerBlock - 1) / edgesPerBlock;
        edge_pass1<<<(int)blocks, E1_WAVES * 32, 0, stream>>>(
            lastu, ei, t, msg, tw, tb, packed, qws, kws,
            ews, logits, m_enc, E);
    }
    {
        long long work = E * 2;
        edge_pass2<<<(int)((work + 255) / 256), 256, 0, stream>>>(
            ei, logits, m_enc, pbuf, denom, E);
    }
    {
        long long work = E * 32;
        edge_pass3<<<(int)((work + 255) / 256), 256, 0, stream>>>(
            ei, ews, vws, pbuf, denom, out, E);
    }
}
